// SpectralNorm_47193100648786
// MI455X (gfx1250) — matrix operand order repaired and checked by reading, not compiled, not run on hardware
//
#include <hip/hip_runtime.h>
#include <stdint.h>

typedef __attribute__((ext_vector_type(16))) __bf16    v16bf;
typedef __attribute__((ext_vector_type(8)))  float     v8f;
typedef __attribute__((ext_vector_type(8)))  uint32_t  v8u;

// Padded NHWC staging plane: [n][y 0..257][x 0..257][hi ci0..31 | lo ci0..31]
// pixel stride = 64 u16 = 128 B.  Interior pixel (y,x) maps to padded (y+1,x+1).
#define PADW   258
#define PIXU16 64            // u16 elements per padded pixel (32 hi + 32 lo)

// ---------- bf16 helpers (round-to-nearest-even split) ----------
__device__ __forceinline__ uint16_t f2bf(float f) {
    uint32_t u = __builtin_bit_cast(uint32_t, f);
    u += 0x7FFFu + ((u >> 16) & 1u);
    return (uint16_t)(u >> 16);
}
__device__ __forceinline__ float bf2f(uint16_t h) {
    uint32_t u = ((uint32_t)h) << 16;
    return __builtin_bit_cast(float, u);
}

// =====================================================================
// Kernel 1: sigma via power iteration on G = W2d*W2d^T (one wave32),
// then pack w_bar/sigma into WMMA A-fragment layout (bf16 hi + lo).
//   g = (((cotile*9 + tap)*2 + part)*32 + lane)*8 + j
// A 16x32 k order used here (B matches; not AMD's table order): half h=lane/16, VGPR j holds K pair
//   k0 = 2j + 16h ; dword = bf16(k0) | bf16(k0+1)<<16
// =====================================================================
__global__ __launch_bounds__(256) void sn_prep_kernel(
        const float* __restrict__ w_bar, uint32_t* __restrict__ wfrag)
{
    __shared__ float Wl[32 * 288];
    __shared__ float Gl[32 * 32];
    __shared__ float s_wscale;
    const int tid = threadIdx.x;

    for (int i = tid; i < 32 * 288; i += 256) Wl[i] = w_bar[i];
    __syncthreads();

    for (int e = tid; e < 1024; e += 256) {
        int i = e >> 5, j = e & 31;
        float s = 0.f;
        for (int k = 0; k < 288; ++k) s += Wl[i * 288 + k] * Wl[j * 288 + k];
        Gl[e] = s;
    }
    __syncthreads();

    if (tid < 32) {
        float v   = 0.17677669529663689f;  // 1/sqrt(32)
        float lam = 1.f;
        for (int it = 0; it < 64; ++it) {
            float u = 0.f;
            #pragma unroll
            for (int j = 0; j < 32; ++j)
                u += Gl[tid * 32 + j] * __shfl(v, j, 32);
            float nrm = u * u;
            #pragma unroll
            for (int o = 16; o > 0; o >>= 1) nrm += __shfl_xor(nrm, o, 32);
            nrm = sqrtf(nrm);
            lam = nrm;                     // ||G v|| -> lambda_max
            v   = u / nrm;
        }
        if (tid == 0) {
            float s0    = sqrtf(lam);
            float sigma = fmaxf(1.f, s0 * 0.5f);
            s_wscale    = 1.f / sigma;
        }
    }
    __syncthreads();
    const float wscale = s_wscale;

    for (int g = tid; g < 9216; g += 256) {
        int j    = g & 7;
        int r    = g >> 3;
        int lane = r & 31;  r >>= 5;
        int part = r & 1;   r >>= 1;
        int t    = r % 9;
        int cot  = r / 9;
        int hh   = lane >> 4;
        int co   = cot * 16 + (lane & 15);
        int k0   = 2 * j + 16 * hh;
        int dh   = t / 3, dw = t % 3;
        float v0 = Wl[((co * 32 + k0    ) * 3 + dh) * 3 + dw] * wscale;
        float v1 = Wl[((co * 32 + k0 + 1) * 3 + dh) * 3 + dw] * wscale;
        uint32_t e0, e1;
        uint16_t h0 = f2bf(v0), h1 = f2bf(v1);
        if (part == 0) { e0 = h0; e1 = h1; }
        else { e0 = f2bf(v0 - bf2f(h0)); e1 = f2bf(v1 - bf2f(h1)); }
        wfrag[g] = e0 | (e1 << 16);
    }
}

// =====================================================================
// Kernel 2a: zero the 1-pixel halo of the padded plane.
// Halo pixels / image: 2*258 (top+bottom rows) + 2*256 (side cols) = 1028.
// Total u32 to clear: 32 * 1028 * 32 = 1,052,672  ->  4112 blocks x 256.
// =====================================================================
__global__ __launch_bounds__(256) void sn_halo_kernel(uint32_t* __restrict__ plane32)
{
    long long i = (long long)blockIdx.x * 256 + threadIdx.x;
    int e = (int)(i & 31);          // u32 within pixel
    long long p = i >> 5;           // halo pixel id, 0..32895
    int n = (int)(p / 1028);
    int r = (int)(p % 1028);
    int y, x;
    if (r < 258)      { y = 0;   x = r;             }
    else if (r < 516) { y = 257; x = r - 258;       }
    else if (r < 772) { x = 0;   y = r - 516 + 1;   }
    else              { x = 257; y = r - 772 + 1;   }
    long long pp = ((long long)n * PADW + y) * PADW + x;
    plane32[pp * 32 + e] = 0u;
}

// =====================================================================
// Kernel 2b: NCHW f32 -> padded interleaved NHWC bf16 hi/lo plane.
// LDS transpose tile of 32 ci x 64 pixels; packed b32 stores.
// =====================================================================
__global__ __launch_bounds__(256) void sn_cvt_kernel(
        const float* __restrict__ x, uint32_t* __restrict__ plane32)
{
    __shared__ float tile[32][65];
    const int b  = blockIdx.x;
    const int n  = b >> 10;                          // 1024 blocks / image
    const int p0 = (b & 1023) << 6;                  // base local pixel
    const int tid = threadIdx.x;

    #pragma unroll
    for (int k = 0; k < 8; ++k) {
        int idx = tid + k * 256;
        int ci  = idx >> 6;
        int col = idx & 63;
        tile[ci][col] = x[((long long)(n * 32 + ci)) * 65536 + p0 + col];
    }
    __syncthreads();
    #pragma unroll
    for (int k = 0; k < 4; ++k) {                    // 1024 ci-pairs
        int idx = tid + k * 256;
        int col = idx >> 4;                          // pixel 0..63
        int cp  = idx & 15;                          // ci pair 0..15
        float v0 = tile[2 * cp    ][col];
        float v1 = tile[2 * cp + 1][col];
        uint16_t h0 = f2bf(v0), h1 = f2bf(v1);
        uint16_t l0 = f2bf(v0 - bf2f(h0));
        uint16_t l1 = f2bf(v1 - bf2f(h1));
        int pl = p0 + col;
        int y  = pl >> 8, xc = pl & 255;
        long long pp = ((long long)n * PADW + (y + 1)) * PADW + (xc + 1);
        uint32_t* pw = plane32 + pp * 32;
        pw[cp]      = (uint32_t)h0 | ((uint32_t)h1 << 16);
        pw[16 + cp] = (uint32_t)l0 | ((uint32_t)l1 << 16);
    }
}

// =====================================================================
// Kernel 3: implicit-GEMM conv, v_wmma_f32_16x16x32_bf16, bf16x3.
// Wave = 16 consecutive x-pixels x 32 out-channels; 9 taps are pure
// immediate-offset loads off one per-lane base (halo supplies zeros).
// 54 WMMAs / wave-tile.
// =====================================================================
__global__ __launch_bounds__(256) void sn_conv_kernel(
        const uint16_t* __restrict__ plane, const uint32_t* __restrict__ wfrag,
        const float* __restrict__ bias, float* __restrict__ out)
{
    const int tid  = threadIdx.x;
    const int lane = tid & 31;
    const int wave = tid >> 5;
    const int wt   = blockIdx.x * 8 + wave;   // 131072 wave-tiles
    const int n    = wt >> 12;                // 4096 tiles / image
    const int rem  = wt & 4095;
    const int y    = rem >> 4;
    const int x0   = (rem & 15) << 4;
    const int hh   = lane >> 4;
    const int pxl  = lane & 15;               // B column = pixel in tile
    const int ciof = hh << 4;                 // lanes 16-31: ci 16..31

    // tap(0,0) reads padded (y, x0+pxl): unpadded (y-1, x-1) + halo shift
    const uint16_t* base =
        plane + (((long long)n * PADW + y) * PADW + (x0 + pxl)) * PIXU16 + ciof;

    v8f acc0 = {};
    v8f acc1 = {};
    const v16bf* wf = (const v16bf*)wfrag;

    #pragma unroll
    for (int t = 0; t < 9; ++t) {
        const int off = ((t / 3) * PADW + (t % 3)) * PIXU16; // const per iter
        v8u bhu = *(const v8u*)(base + off);
        v8u blu = *(const v8u*)(base + off + 32);
        v16bf bh = __builtin_bit_cast(v16bf, bhu);
        v16bf bl = __builtin_bit_cast(v16bf, blu);

        v16bf ah0 = wf[((0 * 9 + t) * 2 + 0) * 32 + lane];
        v16bf al0 = wf[((0 * 9 + t) * 2 + 1) * 32 + lane];
        v16bf ah1 = wf[((1 * 9 + t) * 2 + 0) * 32 + lane];
        v16bf al1 = wf[((1 * 9 + t) * 2 + 1) * 32 + lane];

        acc0 = __builtin_amdgcn_wmma_f32_16x16x32_bf16(false, ah0, false, bh,
                                                       (short)0, acc0, false, false);
        acc0 = __builtin_amdgcn_wmma_f32_16x16x32_bf16(false, ah0, false, bl,
                                                       (short)0, acc0, false, false);
        acc0 = __builtin_amdgcn_wmma_f32_16x16x32_bf16(false, al0, false, bh,
                                                       (short)0, acc0, false, false);
        acc1 = __builtin_amdgcn_wmma_f32_16x16x32_bf16(false, ah1, false, bh,
                                                       (short)0, acc1, false, false);
        acc1 = __builtin_amdgcn_wmma_f32_16x16x32_bf16(false, ah1, false, bl,
                                                       (short)0, acc1, false, false);
        acc1 = __builtin_amdgcn_wmma_f32_16x16x32_bf16(false, al1, false, bh,
                                                       (short)0, acc1, false, false);
    }

    // D layout: VGPR r -> M(co) = r + 8*hh ; lane%16 -> N(pixel)
    const long long pix = (long long)y * 256 + x0 + pxl;
    #pragma unroll
    for (int r = 0; r < 8; ++r) {
        int co0 = r + 8 * hh;
        int co1 = co0 + 16;
        out[((long long)(n * 32 + co0)) * 65536 + pix] = acc0[r] + bias[co0];
        out[((long long)(n * 32 + co1)) * 65536 + pix] = acc1[r] + bias[co1];
    }
}

// =====================================================================
extern "C" void kernel_launch(void* const* d_in, const int* in_sizes, int n_in,
                              void* d_out, int out_size, void* d_ws, size_t ws_size,
                              hipStream_t stream)
{
    const float* x     = (const float*)d_in[0];   // (32,32,256,256)
    const float* w_bar = (const float*)d_in[1];   // (32,32,3,3)
    const float* bias  = (const float*)d_in[2];   // (32,)
    float* out = (float*)d_out;

    char* ws = (char*)d_ws;
    // padded plane: 32 * 258 * 258 * 128 B = 272,646,144 B
    uint32_t* plane32 = (uint32_t*)ws;
    uint32_t* wfrag   = (uint32_t*)(ws + 272646144LL);   // 36,864 B

    sn_prep_kernel<<<1, 256, 0, stream>>>(w_bar, wfrag);
    sn_halo_kernel<<<4112, 256, 0, stream>>>(plane32);
    sn_cvt_kernel<<<32768, 256, 0, stream>>>(x, plane32);
    sn_conv_kernel<<<16384, 256, 0, stream>>>((const uint16_t*)plane32, wfrag,
                                              bias, out);
}
